// naive_nn_14113262534838
// MI455X (gfx1250) — matrix operand order repaired and checked by reading, not compiled, not run on hardware
//
#include <hip/hip_runtime.h>
#include <hip/hip_fp16.h>

typedef __attribute__((ext_vector_type(16))) _Float16 v16h;
typedef __attribute__((ext_vector_type(8)))  float    v8f;

#define BQ     8192
#define TTQ    30
#define METAQ  128
#define HSZ    256
#define HIDQ   512
#define OUTQ   23
#define LQ     5
#define TPRED  90
#define NSTATE 16

__device__ __forceinline__ float sigmoidf_(float x) { return 1.f / (1.f + __expf(-x)); }

// ---------------------------------------------------------------------------
// Generic f32 -> f16 cast with column slice + zero pad.
// ---------------------------------------------------------------------------
__global__ void cast_f16_kernel(_Float16* __restrict__ dst, const float* __restrict__ src,
                                int rows, int scols, int dcols, int src_ld, int src_off) {
  int i = blockIdx.x * blockDim.x + threadIdx.x;
  int total = rows * dcols;
  if (i >= total) return;
  int r = i / dcols, c = i % dcols;
  float v = (c < scols) ? src[(size_t)r * src_ld + src_off + c] : 0.f;
  dst[i] = (_Float16)v;
}

// Layer-0 input: ts = model_input[:, :30] / population, padded to 32 cols (stride 256)
__global__ void prep_x0_kernel(_Float16* __restrict__ x16, const float* __restrict__ mi,
                               const float* __restrict__ pop) {
  int i = blockIdx.x * blockDim.x + threadIdx.x;
  if (i >= BQ * 32) return;
  int b = i >> 5, c = i & 31;
  float v = (c < TTQ) ? mi[(size_t)b * (TTQ + METAQ) + c] / pop[b] : 0.f;
  x16[(size_t)b * HSZ + c] = (_Float16)v;
}

// ---------------------------------------------------------------------------
// Dual-row WMMA accumulation, rotated software pipeline (no in-loop guards):
//   acc0 += A[mBase..+15, :] @ W^T tile;  acc1 += A[mBase+16..+31, :] @ W^T tile
// B fragment shared by both WMMAs; loads for step k issue before the WMMAs of
// step k-32; final step drains outside the loop. K must be a multiple of 32.
// ---------------------------------------------------------------------------
template <bool NCHECK>
__device__ __forceinline__ void wmma_acc_dual(v8f& acc0, v8f& acc1,
    const _Float16* __restrict__ A, int lda,
    const _Float16* __restrict__ W, int ldw,
    int K, int mBase, int nBase, int Nvalid, int lane) {
  int lhi = lane >> 4;
  int lo  = lane & 15;
  const _Float16* Arow0 = A + (size_t)(mBase + lo) * lda;      // A: M = lane%16
  const _Float16* Arow1 = Arow0 + (size_t)16 * lda;            // second m-tile
  int nrow = nBase + lo;                                        // B: N = lane%16
  bool nOK = NCHECK ? (nrow < Nvalid) : true;
  const _Float16* Wrow = W + (size_t)((NCHECK && !nOK) ? 0 : nrow) * ldw;

  auto loadA = [&](const _Float16* Arow, int k0) {
    v16h a;
#pragma unroll
    for (int h = 0; h < 16; ++h)
      a[h] = Arow[k0 + ((h < 8) ? h : (h + 8)) + (lhi << 3)];  // 16x32 f16 A layout
    return a;
  };
  auto loadB = [&](int k0) {
    v16h bm;
#pragma unroll
    for (int h = 0; h < 16; ++h) {                             // 32x16 f16 B layout
      _Float16 wv = Wrow[k0 + ((h < 8) ? h : (h + 8)) + (lhi << 3)];
      bm[h] = NCHECK ? (nOK ? wv : (_Float16)0.f) : wv;
    }
    return bm;
  };

  v16h a0c = loadA(Arow0, 0);
  v16h a1c = loadA(Arow1, 0);
  v16h bc  = loadB(0);
  for (int k0 = 32; k0 < K; k0 += 32) {
    __builtin_prefetch(Arow0 + k0 + 32, 0, 1);   // speculative global_prefetch_b8
    v16h a0n = loadA(Arow0, k0);
    v16h a1n = loadA(Arow1, k0);
    v16h bn  = loadB(k0);
    acc0 = __builtin_amdgcn_wmma_f32_16x16x32_f16(false, a0c, false, bc, (short)0, acc0, false, false);
    acc1 = __builtin_amdgcn_wmma_f32_16x16x32_f16(false, a1c, false, bc, (short)0, acc1, false, false);
    a0c = a0n; a1c = a1n; bc = bn;
  }
  acc0 = __builtin_amdgcn_wmma_f32_16x16x32_f16(false, a0c, false, bc, (short)0, acc0, false, false);
  acc1 = __builtin_amdgcn_wmma_f32_16x16x32_f16(false, a1c, false, bc, (short)0, acc1, false, false);
}

// ---------------------------------------------------------------------------
// Fused LSTM layer: 4 waves/block, one per gate (i,f,g,o) of a 32x16 patch
// (32 batch rows x 16 hidden units); gate tiles exchanged via 8KB LDS,
// nonlinearity applied in-block. No f32 gate matrix ever touches HBM.
// ---------------------------------------------------------------------------
__global__ void lstm_fused_kernel(
    const _Float16* __restrict__ xin, int K1, const _Float16* __restrict__ Wih, int ldw1,
    const _Float16* __restrict__ hin, const _Float16* __restrict__ Whh,
    const float* __restrict__ bih, const float* __restrict__ bhh,
    const float* __restrict__ c0l,
    _Float16* __restrict__ xout, _Float16* __restrict__ dcat) {
  __shared__ float gtile[4][32][16];
  int wave = threadIdx.x >> 5;
  int lane = threadIdx.x & 31;
  int mBase = blockIdx.x * 32;
  int jBase = blockIdx.y * 16;
  int nBase = wave * HSZ + jBase;

  v8f acc0 = {}, acc1 = {};
  wmma_acc_dual<false>(acc0, acc1, xin, HSZ, Wih, ldw1, K1, mBase, nBase, 4 * HSZ, lane);
  wmma_acc_dual<false>(acc0, acc1, hin, HSZ, Whh, HSZ, HSZ, mBase, nBase, 4 * HSZ, lane);

  int lhi = lane >> 4, lo = lane & 15;
  int n = nBase + lo;
  float bv = bih[n] + bhh[n];
#pragma unroll
  for (int r = 0; r < 8; ++r) {
    gtile[wave][r + (lhi << 3)][lo]      = acc0[r] + bv;   // rows 0..15
    gtile[wave][16 + r + (lhi << 3)][lo] = acc1[r] + bv;   // rows 16..31
  }
  __syncthreads();

  // 128 threads apply the gate nonlinearity to 512 elements (4 each, strided)
#pragma unroll
  for (int e = 0; e < 4; ++e) {
    int ii = threadIdx.x + 128 * e;     // 0..511
    int row = ii >> 4, jj = ii & 15;
    float gi = gtile[0][row][jj], gf = gtile[1][row][jj];
    float gg = gtile[2][row][jj], go = gtile[3][row][jj];
    int b = mBase + row, j = jBase + jj;
    float c = sigmoidf_(gf) * c0l[(size_t)b * HSZ + j] + sigmoidf_(gi) * tanhf(gg);
    float xv = sigmoidf_(go) * tanhf(c);
    xout[(size_t)b * HSZ + j] = (_Float16)xv;
    if (dcat) dcat[(size_t)b * HIDQ + j] = (_Float16)xv;
  }
}

// ---------------------------------------------------------------------------
// Generic WMMA GEMM for the MLP stages (dual m-tile per wave).
// ---------------------------------------------------------------------------
__global__ void wmma_gemm_kernel(
    const _Float16* __restrict__ A1, int lda1, const _Float16* __restrict__ W1, int ldw1, int K1,
    const float* __restrict__ bias1,
    float* __restrict__ out32, int ldo32,
    _Float16* __restrict__ out16, int ldo16,
    int Nvalid, int nTiles, int act,
    const float* __restrict__ range, const int* __restrict__ sigscale) {
  int wave = threadIdx.x >> 5;
  int lane = threadIdx.x & 31;
  int mBase = blockIdx.x * 32;
  int nt = blockIdx.y * 4 + wave;
  if (nt >= nTiles) return;           // wave-uniform branch: EXEC stays all-ones
  int nBase = nt * 16;

  v8f acc0 = {}, acc1 = {};
  wmma_acc_dual<true>(acc0, acc1, A1, lda1, W1, ldw1, K1, mBase, nBase, Nvalid, lane);

  int lhi = lane >> 4, lo = lane & 15;
  int n = nBase + lo;
  bool nOK = n < Nvalid;
  float bv = 0.f;
  if (nOK && bias1) bv = bias1[n];
  float sc = 1.f, rg = 1.f;
  if (act == 3) {
    sc = (float)(sigscale ? sigscale[0] : 1);
    rg = (nOK && range) ? range[n] : 1.f;
  }
#pragma unroll
  for (int half = 0; half < 2; ++half) {
    v8f acc = half ? acc1 : acc0;
#pragma unroll
    for (int r = 0; r < 8; ++r) {
      int m = mBase + 16 * half + r + (lhi << 3);
      float v = acc[r] + bv;
      if (act == 1)      v = fmaxf(v, 0.f);
      else if (act == 2) v = tanhf(v);
      else if (act == 3) v = sigmoidf_(v * sc) * rg;
      if (nOK) {
        if (out32) out32[(size_t)m * ldo32 + n] = v;
        if (out16) out16[(size_t)m * ldo16 + n] = (_Float16)v;
      }
    }
  }
}

// ---------------------------------------------------------------------------
// DELPHI ODE, lane-parallel: 16 lanes per batch row (state = lane%16, two rows
// per wave32). Cross-state coupling via intra-16-lane shuffles. RK4, dt=0.25,
// gamma(t)/p_dth_mod(t) hoisted (k2/k3 share the midpoint t). 4096 waves.
// ---------------------------------------------------------------------------
#define R_D   0.34657359028f   // ln2/2
#define R_I   0.13862943611f   // ln2/5
#define R_RI  0.06931471806f   // ln2/10
#define R_RH  0.04620981204f   // ln2/15
#define R_RV  0.06931471806f   // ln2/10
#define PIF   3.14159265359f

__device__ __forceinline__ float shfl16(float v, int src) { return __shfl(v, src, 16); }

__global__ void ode_lane_kernel(float* __restrict__ sol, const float* __restrict__ dnn,
                                const float* __restrict__ pop, const float* __restrict__ mort,
                                const float* __restrict__ cc) {
  // dy = det*(A + B*pdm) + (DC - ISDTH*r_dth)*y + extra
  static const float A_TAB[16]  = {0.f, 0.f, -1.f, 0.8f, 0.006f, 0.194f, 0.f, 0.f, 0.f,
                                   0.f, 0.f, 0.006f, 0.0015f, 0.f, 0.f, 0.2f};
  static const float B_TAB[16]  = {0.f, 0.f, 0.f, -0.8f, -0.006f, -0.194f, 0.8f, 0.006f, 0.194f,
                                   0.f, 0.f, 0.f, -0.0015f, 0.0015f, 0.f, 0.f};
  static const float DC_TAB[16] = {0.f, 0.f, 0.f, -R_RI, -R_RH, -R_RI, 0.f, 0.f, 0.f,
                                   0.f, 0.f, 0.f, -R_RV, 0.f, 0.f, 0.f};
  static const float DTH_TAB[16] = {0.f, 0.f, 0.f, 0.f, 0.f, 0.f, 1.f, 1.f, 1.f,
                                    0.f, 0.f, 0.f, 0.f, 1.f, 0.f, 0.f};

  int tid = blockIdx.x * blockDim.x + threadIdx.x;
  int b   = tid >> 4;
  int sub = tid & 15;
  if (b >= BQ) return;

  const float* d = dnn + (size_t)b * OUTQ;
  float alpha = d[11], days = d[12], r_s = d[13], r_dth = d[14], p_dth = d[15];
  float r_dthdecay = d[16], k1p = d[17], k2p = d[18], jump = d[19], t_jump = d[20];
  float stdn = d[21], k3p = d[22];
  float n = pop[b];
  float inv_n = 1.f / n;

  float PopD = floorf(mort[b] * cc[b]);
  float PopI = cc[b];
  float R0 = (PopI - PopD > 5.f * PopD) ? 5.f * PopD : 0.f;
  float PopR = fminf(PopI - PopD - 1.f, fminf(floorf(R0), 10.f));
  float CI = (PopI - PopD - PopR) * k3p;
  float Cd = CI * 5.f;

  float y =
      (sub == 0)  ? n - Cd - Cd * (k1p + k2p) - PopR * 5.f - PopD * 5.f :
      (sub == 1)  ? Cd * k1p :
      (sub == 2)  ? Cd * k2p :
      (sub == 3)  ? (Cd - CI) * (1.f - p_dth) :
      (sub == 4)  ? CI * 0.03f * (1.f - p_dth) :
      (sub == 5)  ? CI * 0.97f * (1.f - p_dth) :
      (sub == 6)  ? (Cd - CI) * p_dth :
      (sub == 7)  ? CI * 0.03f * p_dth :
      (sub == 8)  ? CI * 0.97f * p_dth :
      (sub == 9)  ? PopR * 5.f :
      (sub == 10) ? PopD * 5.f :
      (sub == 11) ? CI * 0.03f :
      (sub == 12) ? CI * 0.0075f * (1.f - p_dth) :
      (sub == 13) ? CI * 0.0075f * p_dth :
      (sub == 14) ? PopD : PopI;

  float aS = A_TAB[sub], bS = B_TAB[sub];
  float dcS = DC_TAB[sub] - DTH_TAB[sub] * r_dth;

  float* out = sol + (size_t)b * TPRED * NSTATE;
  out[sub] = y;

  auto gfun = [&](float t, float& gamma, float& pdm) {
    float dtj = t - t_jump;
    gamma = (2.f / PIF) * atanf(-(t - days) * 0.05f * r_s) + 1.f
          + jump * __expf(-(dtj * dtj) / (2.f * stdn * stdn));
    pdm = (2.f / PIF) * (p_dth - 0.001f) * (atanf(-t * 0.05f * r_dthdecay) + PIF * 0.5f) + 0.001f;
  };

  auto deriv = [&](float yv, float gamma, float pdm) -> float {
    float S  = shfl16(yv, 0), E  = shfl16(yv, 1), I  = shfl16(yv, 2);
    float v3 = shfl16(yv, 3), v4 = shfl16(yv, 4), v5 = shfl16(yv, 5);
    float v6 = shfl16(yv, 6), v7 = shfl16(yv, 7), v8 = shfl16(yv, 8);
    float new_inf = alpha * gamma * S * I * inv_n;
    float det = R_D * I;
    float riE = R_I * E;
    float extra =
        (sub == 0)  ? -new_inf :
        (sub == 1)  ? (new_inf - riE) :
        (sub == 2)  ? riE :
        (sub == 9)  ? (R_RI * (v3 + v5) + R_RH * v4) :
        (sub == 10) ? (r_dth * (v6 + v7 + v8)) :
        (sub == 14) ? (r_dth * (v7 + v8)) : 0.f;
    return det * (aS + bS * pdm) + dcS * yv + extra;
  };

  const int SUB = 4;
  const float dtf = 1.f / (float)SUB;
  const float hdt = 0.5f * dtf;
  float t = 0.f;
  for (int ti = 1; ti < TPRED; ++ti) {
    for (int ss = 0; ss < SUB; ++ss) {
      float g0, pd0, g1, pd1, g2, pd2;
      gfun(t, g0, pd0);
      gfun(t + hdt, g1, pd1);
      gfun(t + dtf, g2, pd2);
      float q1 = deriv(y, g0, pd0);
      float q2 = deriv(y + hdt * q1, g1, pd1);
      float q3 = deriv(y + hdt * q2, g1, pd1);
      float q4 = deriv(y + dtf * q3, g2, pd2);
      y += (dtf / 6.f) * (q1 + 2.f * q2 + 2.f * q3 + q4);
      t += dtf;
    }
    out[(size_t)ti * NSTATE + sub] = y;
  }
}

// ---------------------------------------------------------------------------
// Host driver
// ---------------------------------------------------------------------------
extern "C" void kernel_launch(void* const* d_in, const int* in_sizes, int n_in,
                              void* d_out, int out_size, void* d_ws, size_t ws_size,
                              hipStream_t stream) {
  const float* mi    = (const float*)d_in[0];
  const float* pop   = (const float*)d_in[1];
  const float* mort  = (const float*)d_in[2];
  const float* cc    = (const float*)d_in[3];
  const float* h0    = (const float*)d_in[4];
  const float* c0    = (const float*)d_in[5];
  const float* W_ih0 = (const float*)d_in[6];
  const float* W_ih  = (const float*)d_in[7];
  const float* W_hh  = (const float*)d_in[8];
  const float* b_ih  = (const float*)d_in[9];
  const float* b_hh  = (const float*)d_in[10];
  const float* Wm1   = (const float*)d_in[11];
  const float* bm1   = (const float*)d_in[12];
  const float* Wm2   = (const float*)d_in[13];
  const float* bm2   = (const float*)d_in[14];
  const float* Wr1   = (const float*)d_in[15];
  const float* br1   = (const float*)d_in[16];
  const float* Wr2   = (const float*)d_in[17];
  const float* br2   = (const float*)d_in[18];
  const float* rng   = (const float*)d_in[19];
  const int*   sscal = (const int*)d_in[20];
  (void)in_sizes; (void)n_in; (void)out_size; (void)ws_size;

  char* ws = (char*)d_ws;
  size_t off = 0;
  auto alloc = [&](size_t bytes) -> char* {
    char* ptr = ws + off;
    off = (off + bytes + 255) & ~(size_t)255;
    return ptr;
  };
  _Float16* x16a    = (_Float16*)alloc((size_t)BQ * HSZ * 2);
  _Float16* x16b    = (_Float16*)alloc((size_t)BQ * HSZ * 2);
  _Float16* h16     = (_Float16*)alloc((size_t)BQ * HSZ * 2);
  _Float16* min16   = (_Float16*)alloc((size_t)BQ * METAQ * 2);
  _Float16* m116    = (_Float16*)alloc((size_t)BQ * HSZ * 2);
  _Float16* dcat16  = (_Float16*)alloc((size_t)BQ * HIDQ * 2);
  _Float16* r116    = (_Float16*)alloc((size_t)BQ * HIDQ * 2);
  _Float16* wih0_16 = (_Float16*)alloc((size_t)1024 * 32 * 2);
  _Float16* wih16   = (_Float16*)alloc((size_t)4 * 1024 * 256 * 2);
  _Float16* whh16   = (_Float16*)alloc((size_t)5 * 1024 * 256 * 2);
  _Float16* wm1_16  = (_Float16*)alloc((size_t)256 * 128 * 2);
  _Float16* wm2_16  = (_Float16*)alloc((size_t)256 * 256 * 2);
  _Float16* wr1_16  = (_Float16*)alloc((size_t)512 * 512 * 2);
  _Float16* wr2_16  = (_Float16*)alloc((size_t)23 * 512 * 2);

  auto blocks = [](size_t nthreads) { return dim3((unsigned)((nthreads + 255) / 256)); };

  // Weight casts (f32 -> f16, layer-0 K padded 30 -> 32)
  cast_f16_kernel<<<blocks((size_t)1024 * 32), 256, 0, stream>>>(wih0_16, W_ih0, 1024, 30, 32, 30, 0);
  cast_f16_kernel<<<blocks((size_t)4096 * 256), 256, 0, stream>>>(wih16, W_ih, 4096, 256, 256, 256, 0);
  cast_f16_kernel<<<blocks((size_t)5120 * 256), 256, 0, stream>>>(whh16, W_hh, 5120, 256, 256, 256, 0);
  cast_f16_kernel<<<blocks((size_t)256 * 128), 256, 0, stream>>>(wm1_16, Wm1, 256, 128, 128, 128, 0);
  cast_f16_kernel<<<blocks((size_t)256 * 256), 256, 0, stream>>>(wm2_16, Wm2, 256, 256, 256, 256, 0);
  cast_f16_kernel<<<blocks((size_t)512 * 512), 256, 0, stream>>>(wr1_16, Wr1, 512, 512, 512, 512, 0);
  cast_f16_kernel<<<blocks((size_t)23 * 512), 256, 0, stream>>>(wr2_16, Wr2, 23, 512, 512, 512, 0);

  // Activation casts
  prep_x0_kernel<<<blocks((size_t)BQ * 32), 256, 0, stream>>>(x16a, mi, pop);
  cast_f16_kernel<<<blocks((size_t)BQ * METAQ), 256, 0, stream>>>(min16, mi, BQ, METAQ, METAQ, TTQ + METAQ, TTQ);

  float* dnn = (float*)d_out + (size_t)BQ * TPRED * NSTATE;

  // LSTM stack: fused GEMM + gates, double-buffered activations
  _Float16* xcur = x16a;
  _Float16* xnxt = x16b;
  for (int l = 0; l < LQ; ++l) {
    cast_f16_kernel<<<blocks((size_t)BQ * HSZ), 256, 0, stream>>>(
        h16, h0 + (size_t)l * BQ * HSZ, BQ, HSZ, HSZ, HSZ, 0);
    const _Float16* W1 = (l == 0) ? wih0_16 : (wih16 + (size_t)(l - 1) * 1024 * 256);
    int ldw1 = (l == 0) ? 32 : 256;
    int K1   = (l == 0) ? 32 : 256;
    dim3 grid(BQ / 32, HSZ / 16);     // (256, 16): one block per 32x16 patch
    lstm_fused_kernel<<<grid, 128, 0, stream>>>(
        xcur, K1, W1, ldw1,
        h16, whh16 + (size_t)l * 1024 * 256,
        b_ih + (size_t)l * 1024, b_hh + (size_t)l * 1024,
        c0 + (size_t)l * BQ * HSZ,
        xnxt, (l == LQ - 1) ? dcat16 : nullptr);
    _Float16* tmp = xcur; xcur = xnxt; xnxt = tmp;
  }

  // Meta MLP: relu(minp@Wm1^T + bm1), relu(@Wm2^T + bm2) -> dcat[:, 256:512]
  {
    dim3 grid(BQ / 32, 16 / 4);
    wmma_gemm_kernel<<<grid, 128, 0, stream>>>(
        min16, METAQ, wm1_16, METAQ, METAQ,
        bm1, nullptr, 0, m116, HSZ,
        HSZ, 16, 1, nullptr, nullptr);
    wmma_gemm_kernel<<<grid, 128, 0, stream>>>(
        m116, HSZ, wm2_16, HSZ, HSZ,
        bm2, nullptr, 0, dcat16 + HSZ, HIDQ,
        HSZ, 16, 1, nullptr, nullptr);
  }

  // Readout: r1 = tanh(dcat@Wr1^T + br1); dnn = sigmoid((r1@Wr2^T + br2)*s) * range
  {
    dim3 grid(BQ / 32, 32 / 4);
    wmma_gemm_kernel<<<grid, 128, 0, stream>>>(
        dcat16, HIDQ, wr1_16, HIDQ, HIDQ,
        br1, nullptr, 0, r116, HIDQ,
        HIDQ, 32, 2, nullptr, nullptr);
    dim3 grid2(BQ / 32, 1);           // N=23 -> 2 n-tiles (waves 2,3 idle)
    wmma_gemm_kernel<<<grid2, 128, 0, stream>>>(
        r116, HIDQ, wr2_16, HIDQ, HIDQ,
        br2, dnn, OUTQ, nullptr, 0,
        OUTQ, 2, 3, rng, sscal);
  }

  // ODE integration (lane-parallel) -> sol at d_out[0 .. B*90*16)
  ode_lane_kernel<<<dim3((BQ * 16) / 256), 256, 0, stream>>>((float*)d_out, dnn, pop, mort, cc);
}
